// Mamba2LMHeadModel_34969623724561
// MI455X (gfx1250) — matrix operand order repaired and checked by reading, not compiled, not run on hardware
//
#include <hip/hip_runtime.h>
#include <hip/hip_bf16.h>
#include <math.h>

#define D_MODEL 768
#define N_LAYER 4
#define D_STATE 128
#define HEADDIM 64
#define VOCAB 50288
#define VOCAB_PAD 50304
#define D_INNER 1536
#define NHEADS 24
#define CONV_CH 1792
#define D_IN_PROJ 3352
#define D_IN_PROJ_PAD 3392
#define SEQ 2048
#define NCHUNKS 32
#define EPS 1e-5f

typedef __bf16 bf16;
typedef __attribute__((ext_vector_type(16))) __bf16 v16bf;
typedef __attribute__((ext_vector_type(8)))  float  v8f;

// ---------------------------------------------------------------------------
// WMMA helpers (wave32, 16x16x32 bf16 -> f32)
// ---------------------------------------------------------------------------
__device__ __forceinline__ v8f wmma_bf16(v16bf a, v16bf b, v8f c) {
  return __builtin_amdgcn_wmma_f32_16x16x32_bf16(false, a, false, b, (short)0, c,
                                                 false, false);
}

// A fragment: 16x32 bf16, row-major in LDS with pitch (elements).
// lanes 0-15: row=lane, K {0..7,16..23}; lanes 16-31: row=lane-16, K {8..15,24..31}
__device__ __forceinline__ v16bf frag_a(const bf16* base, int pitch, int lane) {
  int r  = lane & 15;
  int k0 = (lane & 16) ? 8 : 0;
  const bf16* p = base + r * pitch + k0;
  v16bf v;
  ((uint4*)&v)[0] = *(const uint4*)p;
  ((uint4*)&v)[1] = *(const uint4*)(p + 16);
  return v;
}

// B fragment from LDS staged as Bs[N][K] (i.e. B transposed), pitch in elements.
// lanes 0-15: col=lane, K {0..7,16..23}; lanes 16-31: col=lane-16, K {8..15,24..31}
__device__ __forceinline__ v16bf frag_b(const bf16* base, int pitch, int lane) {
  int c  = lane & 15;
  int k0 = (lane & 16) ? 8 : 0;
  const bf16* p = base + c * pitch + k0;
  v16bf v;
  ((uint4*)&v)[0] = *(const uint4*)p;
  ((uint4*)&v)[1] = *(const uint4*)(p + 16);
  return v;
}

// ---------------------------------------------------------------------------
// CDNA5 async global->LDS staging (tracked by ASYNCcnt).
// Generic pointers to __shared__ carry the LDS byte address in their low
// 32 bits.
// ---------------------------------------------------------------------------
__device__ __forceinline__ unsigned lds32(const void* p) {
  return (unsigned)(unsigned long long)p;
}
__device__ __forceinline__ void async_b128(unsigned lds, const void* base,
                                           unsigned voff) {
  asm volatile("global_load_async_to_lds_b128 %0, %1, %2"
               :
               : "v"(lds), "v"(voff), "s"(base)
               : "memory");
}
#define S_WAIT_ASYNC_0() asm volatile("s_wait_asynccnt 0x0" ::: "memory")
#define S_WAIT_ASYNC_2() asm volatile("s_wait_asynccnt 0x2" ::: "memory")
#define S_WAIT_ASYNC_3() asm volatile("s_wait_asynccnt 0x3" ::: "memory")

// ---------------------------------------------------------------------------
// Generic GEMM: C[M,N](f32) = A[M,K](bf16, row-major) x B(bf16)
// BT=0: B is [K, ldb] row-major.  BT=1: B is [N, K] row-major (pre-transposed).
// NT=1: non-temporal C stores (streaming output, e.g. logits).
// Tile: 128(M) x 64(N), 8 waves of 32x32, K step 32, double-buffered async LDS.
// ---------------------------------------------------------------------------
template <int BT, int NT>
__global__ __launch_bounds__(256) void gemm_bf16_kernel(
    const bf16* __restrict__ A, const bf16* __restrict__ B, float* __restrict__ C,
    int K, int ldb, int ldc, int nguard) {
  __shared__ __align__(16) bf16 As[2][128 * 40];
  __shared__ __align__(16) bf16 Bs[2][64 * 40];
  int tid = threadIdx.x, lane = tid & 31, wave = tid >> 5;
  int wm = wave >> 1, wn = wave & 1;
  int bm = blockIdx.x * 128, bn = blockIdx.y * 64;
  v8f a00 = {}, a01 = {}, a10 = {}, a11 = {};

  int ar = tid >> 1, ah = tid & 1;   // A staging: row, 16-elem half
  int br1 = tid >> 2, bq1 = tid & 3; // B staging (BT=1)
  int br0 = tid >> 3, bq0 = tid & 7; // B staging (BT=0)

  auto stage = [&](int k0, int buf) {
    {  // A tile: 128 rows x 32 K, 2 async b128 per thread
      unsigned l0 = lds32(&As[buf][ar * 40 + ah * 16]);
      unsigned go = (unsigned)(((long)(bm + ar) * K + k0 + ah * 16) * 2);
      async_b128(l0, A, go);
      async_b128(l0 + 16, A, go + 16);
    }
    if (BT) {  // B already [N,K]: contiguous along K, 1 async b128 per thread
      unsigned l = lds32(&Bs[buf][br1 * 40 + bq1 * 8]);
      unsigned go = (unsigned)(((long)(bn + br1) * K + k0 + bq1 * 8) * 2);
      async_b128(l, B, go);
    } else {  // B is [K,N]: transpose into Bs[N][K] manually
      const bf16* g = B + (long)(k0 + br0) * ldb + bn + bq0 * 8;
      uint4 u = *(const uint4*)g;
      if (k0 + 32 < K) __builtin_prefetch(g + (long)32 * ldb, 0, 0);
      bf16 tmp[8];
      *(uint4*)tmp = u;
#pragma unroll
      for (int j = 0; j < 8; ++j) Bs[buf][(bq0 * 8 + j) * 40 + br0] = tmp[j];
    }
  };

  const int nk = K / 32;
  stage(0, 0);
  for (int it = 0; it < nk; ++it) {
    int buf = it & 1;
    if (it + 1 < nk) {
      stage((it + 1) * 32, buf ^ 1);
      if (BT) S_WAIT_ASYNC_3();  // tile `it` complete, tile `it+1` in flight
      else    S_WAIT_ASYNC_2();
    } else {
      S_WAIT_ASYNC_0();
    }
    __syncthreads();
    v16bf fa0 = frag_a(&As[buf][(wm * 32) * 40], 40, lane);
    v16bf fa1 = frag_a(&As[buf][(wm * 32 + 16) * 40], 40, lane);
    v16bf fb0 = frag_b(&Bs[buf][(wn * 32) * 40], 40, lane);
    v16bf fb1 = frag_b(&Bs[buf][(wn * 32 + 16) * 40], 40, lane);
    a00 = wmma_bf16(fa0, fb0, a00);
    a01 = wmma_bf16(fa0, fb1, a01);
    a10 = wmma_bf16(fa1, fb0, a10);
    a11 = wmma_bf16(fa1, fb1, a11);
    __syncthreads();
  }

  int m0 = bm + wm * 32, n0 = bn + wn * 32;
  int cr = lane & 15, rb = (lane & 16) ? 8 : 0;
#pragma unroll
  for (int r = 0; r < 8; ++r) {
    int row0 = m0 + rb + r, row1 = m0 + 16 + rb + r;
    int col0 = n0 + cr, col1 = n0 + 16 + cr;
    if (col0 < nguard) {
      if (NT) {
        __builtin_nontemporal_store(a00[r], &C[(long)row0 * ldc + col0]);
        __builtin_nontemporal_store(a10[r], &C[(long)row1 * ldc + col0]);
      } else {
        C[(long)row0 * ldc + col0] = a00[r];
        C[(long)row1 * ldc + col0] = a10[r];
      }
    }
    if (col1 < nguard) {
      if (NT) {
        __builtin_nontemporal_store(a01[r], &C[(long)row0 * ldc + col1]);
        __builtin_nontemporal_store(a11[r], &C[(long)row1 * ldc + col1]);
      } else {
        C[(long)row0 * ldc + col1] = a01[r];
        C[(long)row1 * ldc + col1] = a11[r];
      }
    }
  }
}

// ---------------------------------------------------------------------------
// Elementwise / conversion kernels
// ---------------------------------------------------------------------------
__global__ void convert_pad_kernel(const float* __restrict__ src, bf16* __restrict__ dst,
                                   int R, int Cin, int Rpad, int Cpad) {
  long i = (long)blockIdx.x * 256 + threadIdx.x;
  if (i >= (long)Rpad * Cpad) return;
  int r = (int)(i / Cpad), c = (int)(i % Cpad);
  float v = (r < R && c < Cin) ? src[(long)r * Cin + c] : 0.0f;
  dst[i] = (bf16)v;
}

__global__ void embed_kernel(const int* __restrict__ ids, const float* __restrict__ emb,
                             float* __restrict__ x) {
  int i = blockIdx.x * 256 + threadIdx.x;
  if (i >= SEQ * D_MODEL) return;
  int t = i / D_MODEL, d = i % D_MODEL;
  x[i] = emb[(long)ids[t] * D_MODEL + d];
}

__global__ __launch_bounds__(256) void rmsnorm_bf16_kernel(
    const float* __restrict__ x, const float* __restrict__ w, bf16* __restrict__ out,
    int width) {
  __shared__ float red[256];
  int t = blockIdx.x, tid = threadIdx.x;
  const float* row = x + (long)t * width;
  float s = 0.f;
  for (int d = tid; d < width; d += 256) { float v = row[d]; s += v * v; }
  red[tid] = s;
  __syncthreads();
  for (int o = 128; o > 0; o >>= 1) {
    if (tid < o) red[tid] += red[tid + o];
    __syncthreads();
  }
  float inv = rsqrtf(red[0] / width + EPS);
  for (int d = tid; d < width; d += 256)
    out[(long)t * width + d] = (bf16)(row[d] * inv * w[d]);
}

__global__ void dt_kernel(const float* __restrict__ zx, const float* __restrict__ dt_bias,
                          float* __restrict__ dt) {
  int i = blockIdx.x * 256 + threadIdx.x;
  if (i >= SEQ * NHEADS) return;
  int t = i / NHEADS, h = i % NHEADS;
  float v = zx[(long)t * D_IN_PROJ_PAD + D_INNER + CONV_CH + h] + dt_bias[h];
  dt[i] = (v > 20.f) ? v : log1pf(__expf(v));
}

__global__ void conv_kernel(const float* __restrict__ zx, const float* __restrict__ cw,
                            const float* __restrict__ cb, float* __restrict__ out) {
  int i = blockIdx.x * 256 + threadIdx.x;
  if (i >= SEQ * CONV_CH) return;
  int t = i / CONV_CH, ch = i % CONV_CH;
  float acc = cb[ch];
#pragma unroll
  for (int j = 0; j < 4; ++j) {
    int tt = t - 3 + j;
    if (tt >= 0) acc += cw[ch * 4 + j] * zx[(long)tt * D_IN_PROJ_PAD + D_INNER + ch];
  }
  out[i] = acc / (1.f + __expf(-acc));  // SiLU
}

__global__ void add_kernel(float* __restrict__ x, const float* __restrict__ o, int n) {
  int i = blockIdx.x * 256 + threadIdx.x;
  if (i < n) x[i] += o[i];
}

// ---------------------------------------------------------------------------
// SSD chunk kernel: one (chunk, head) per workgroup. Computes
//   G = C B^T (WMMA, K=128), M = G .* L, Y_diag = M xd (WMMA, K=64),
//   states = (xd*decay)^T B (WMMA, K=64), and stores Acum.
// ---------------------------------------------------------------------------
__global__ __launch_bounds__(256) void ssd_chunk_kernel(
    const float* __restrict__ xBCc, const float* __restrict__ dtbuf,
    const float* __restrict__ A_log, float* __restrict__ Y, float* __restrict__ states,
    float* __restrict__ AcumG) {
  __shared__ __align__(16) bf16 Cbf[64 * 136];
  __shared__ __align__(16) bf16 Bbf[64 * 136];
  __shared__ __align__(16) bf16 BbfT[128 * 72];
  __shared__ __align__(16) bf16 xdT[64 * 72];
  __shared__ __align__(16) bf16 xdTd[64 * 72];
  __shared__ __align__(16) bf16 Mbf[64 * 72];
  __shared__ float Acum[64];
  __shared__ float sdt[64];

  int chunk = blockIdx.x, head = blockIdx.y;
  int tid = threadIdx.x, lane = tid & 31, wave = tid >> 5;
  int t0 = chunk * 64;
  float Ah = -__expf(A_log[head]);

  if (tid < 64) {
    float d = dtbuf[(t0 + tid) * NHEADS + head];
    sdt[tid] = d;
    Acum[tid] = d * Ah;
  }
  for (int i = tid; i < 64 * 128; i += 256) {
    int l = i >> 7, n = i & 127;
    const float* rowp = xBCc + (long)(t0 + l) * CONV_CH;
    float bv = rowp[D_INNER + n];
    float cv = rowp[D_INNER + D_STATE + n];
    Bbf[l * 136 + n] = (bf16)bv;
    BbfT[n * 72 + l] = (bf16)bv;
    Cbf[l * 136 + n] = (bf16)cv;
  }
  __syncthreads();
  if (tid == 0) {
    float run = 0.f;
    for (int l = 0; l < 64; ++l) { run += Acum[l]; Acum[l] = run; }
  }
  __syncthreads();
  float alast = Acum[63];
  for (int i = tid; i < 64 * 64; i += 256) {
    int l = i >> 6, p = i & 63;
    float v = xBCc[(long)(t0 + l) * CONV_CH + head * HEADDIM + p] * sdt[l];
    xdT[p * 72 + l]  = (bf16)v;
    xdTd[p * 72 + l] = (bf16)(v * __expf(alast - Acum[l]));
  }
  if (tid < 64) AcumG[((long)chunk * NHEADS + head) * 64 + tid] = Acum[tid];
  __syncthreads();

  // G = C @ B^T -> masked/decayed into Mbf
  for (int s = 0; s < 2; ++s) {
    int tI = wave * 2 + s;
    int i = tI >> 2, j = tI & 3;
    v8f acc = {};
#pragma unroll
    for (int k0 = 0; k0 < 128; k0 += 32) {
      v16bf a = frag_a(&Cbf[(i * 16) * 136 + k0], 136, lane);
      v16bf b = frag_b(&Bbf[(j * 16) * 136 + k0], 136, lane);
      acc = wmma_bf16(a, b, acc);
    }
    int cr = lane & 15, rb = (lane & 16) ? 8 : 0;
#pragma unroll
    for (int r = 0; r < 8; ++r) {
      int row = i * 16 + rb + r, col = j * 16 + cr;
      float m = (col <= row) ? acc[r] * __expf(Acum[row] - Acum[col]) : 0.0f;
      Mbf[row * 72 + col] = (bf16)m;
    }
  }
  __syncthreads();

  // Y_diag = M @ xd
  for (int s = 0; s < 2; ++s) {
    int tI = wave * 2 + s;
    int i = tI >> 2, j = tI & 3;
    v8f acc = {};
#pragma unroll
    for (int k0 = 0; k0 < 64; k0 += 32) {
      v16bf a = frag_a(&Mbf[(i * 16) * 72 + k0], 72, lane);
      v16bf b = frag_b(&xdT[(j * 16) * 72 + k0], 72, lane);
      acc = wmma_bf16(a, b, acc);
    }
    int cr = lane & 15, rb = (lane & 16) ? 8 : 0;
#pragma unroll
    for (int r = 0; r < 8; ++r) {
      int l = i * 16 + rb + r, p = j * 16 + cr;
      Y[(long)(t0 + l) * D_INNER + head * HEADDIM + p] = acc[r];
    }
  }

  // states[p][n] = sum_l decay(l) xd[l][p] B[l][n]
  for (int s = 0; s < 4; ++s) {
    int tI = wave * 4 + s;
    int i = tI >> 3, j = tI & 7;
    v8f acc = {};
#pragma unroll
    for (int k0 = 0; k0 < 64; k0 += 32) {
      v16bf a = frag_a(&xdTd[(i * 16) * 72 + k0], 72, lane);
      v16bf b = frag_b(&BbfT[(j * 16) * 72 + k0], 72, lane);
      acc = wmma_bf16(a, b, acc);
    }
    int cr = lane & 15, rb = (lane & 16) ? 8 : 0;
#pragma unroll
    for (int r = 0; r < 8; ++r) {
      int p = i * 16 + rb + r, n = j * 16 + cr;
      states[(((long)chunk * NHEADS + head) * 64 + p) * 128 + n] = acc[r];
    }
  }
}

// Sequential inter-chunk state scan (per head).
__global__ __launch_bounds__(256) void ssd_scan_kernel(
    const float* __restrict__ states, const float* __restrict__ AcumG,
    float* __restrict__ Sprefix) {
  int head = blockIdx.x, tid = threadIdx.x;
  float S[32];
#pragma unroll
  for (int i = 0; i < 32; ++i) S[i] = 0.f;
  for (int c = 0; c < NCHUNKS; ++c) {
    long off = ((long)c * NHEADS + head) * 8192 + tid * 32;
#pragma unroll
    for (int i = 0; i < 32; ++i) Sprefix[off + i] = S[i];
    float dec = __expf(AcumG[((long)c * NHEADS + head) * 64 + 63]);
#pragma unroll
    for (int i = 0; i < 32; ++i) S[i] = S[i] * dec + states[off + i];
  }
}

// Y_off = (C @ Sprefix^T) scaled by exp(Acum[l]); accumulated into Y.
__global__ __launch_bounds__(256) void ssd_yoff_kernel(
    const float* __restrict__ xBCc, const float* __restrict__ Sprefix,
    const float* __restrict__ AcumG, float* __restrict__ Y) {
  __shared__ __align__(16) bf16 Cbf[64 * 136];
  __shared__ __align__(16) bf16 Spb[64 * 136];
  __shared__ float sacc[64];
  int chunk = blockIdx.x, head = blockIdx.y;
  int tid = threadIdx.x, lane = tid & 31, wave = tid >> 5;
  int t0 = chunk * 64;
  if (tid < 64) sacc[tid] = AcumG[((long)chunk * NHEADS + head) * 64 + tid];
  for (int i = tid; i < 64 * 128; i += 256) {
    int l = i >> 7, n = i & 127;
    Cbf[l * 136 + n] = (bf16)xBCc[(long)(t0 + l) * CONV_CH + D_INNER + D_STATE + n];
    Spb[l * 136 + n] = (bf16)Sprefix[(((long)chunk * NHEADS + head) * 64 + l) * 128 + n];
  }
  __syncthreads();
  for (int s = 0; s < 2; ++s) {
    int tI = wave * 2 + s;
    int i = tI >> 2, j = tI & 3;
    v8f acc = {};
#pragma unroll
    for (int k0 = 0; k0 < 128; k0 += 32) {
      v16bf a = frag_a(&Cbf[(i * 16) * 136 + k0], 136, lane);
      v16bf b = frag_b(&Spb[(j * 16) * 136 + k0], 136, lane);
      acc = wmma_bf16(a, b, acc);
    }
    int cr = lane & 15, rb = (lane & 16) ? 8 : 0;
#pragma unroll
    for (int r = 0; r < 8; ++r) {
      int l = i * 16 + rb + r, p = j * 16 + cr;
      long idx = (long)(t0 + l) * D_INNER + head * HEADDIM + p;
      Y[idx] += acc[r] * __expf(sacc[l]);
    }
  }
}

// y = rmsnorm((Y + xs*D) * silu(z), norm_w) -> bf16
__global__ __launch_bounds__(256) void gated_norm_kernel(
    const float* __restrict__ Y, const float* __restrict__ xBCc,
    const float* __restrict__ zx, const float* __restrict__ Dskip,
    const float* __restrict__ norm_w, bf16* __restrict__ out) {
  __shared__ float red[256];
  int t = blockIdx.x, tid = threadIdx.x;
  float u[6];
  float s = 0.f;
#pragma unroll
  for (int q = 0; q < 6; ++q) {
    int d = tid + q * 256;
    float ytot = Y[(long)t * D_INNER + d] + xBCc[(long)t * CONV_CH + d] * Dskip[d >> 6];
    float z = zx[(long)t * D_IN_PROJ_PAD + d];
    float v = ytot * z / (1.f + __expf(-z));
    u[q] = v;
    s += v * v;
  }
  red[tid] = s;
  __syncthreads();
  for (int o = 128; o > 0; o >>= 1) {
    if (tid < o) red[tid] += red[tid + o];
    __syncthreads();
  }
  float inv = rsqrtf(red[0] / D_INNER + EPS);
#pragma unroll
  for (int q = 0; q < 6; ++q) {
    int d = tid + q * 256;
    out[(long)t * D_INNER + d] = (bf16)(u[q] * inv * norm_w[d]);
  }
}

// ---------------------------------------------------------------------------
// Host launcher
// ---------------------------------------------------------------------------
static inline int cdiv(long a, long b) { return (int)((a + b - 1) / b); }

extern "C" void kernel_launch(void* const* d_in, const int* in_sizes, int n_in,
                              void* d_out, int out_size, void* d_ws, size_t ws_size,
                              hipStream_t stream) {
  (void)in_sizes; (void)n_in; (void)out_size; (void)ws_size;
  const int*   ids        = (const int*)d_in[0];
  const float* emb        = (const float*)d_in[1];
  const float* ln_w       = (const float*)d_in[2];
  const float* W_in       = (const float*)d_in[3];
  const float* conv_w     = (const float*)d_in[4];
  const float* conv_b     = (const float*)d_in[5];
  const float* dt_bias    = (const float*)d_in[6];
  const float* A_log      = (const float*)d_in[7];
  const float* D_skip     = (const float*)d_in[8];
  const float* norm_w     = (const float*)d_in[9];
  const float* W_out      = (const float*)d_in[10];
  const float* final_ln_w = (const float*)d_in[11];

  char* ws = (char*)d_ws;
  size_t off = 0;
  auto alloc = [&](size_t bytes) {
    void* p = ws + off;
    off += (bytes + 255) & ~(size_t)255;
    return p;
  };
  float* x       = (float*)alloc((size_t)SEQ * D_MODEL * 4);
  float* zx      = (float*)alloc((size_t)SEQ * D_IN_PROJ_PAD * 4);
  float* dtb     = (float*)alloc((size_t)SEQ * NHEADS * 4);
  float* xBCc    = (float*)alloc((size_t)SEQ * CONV_CH * 4);
  float* Y       = (float*)alloc((size_t)SEQ * D_INNER * 4);
  float* O       = (float*)alloc((size_t)SEQ * D_MODEL * 4);
  float* states  = (float*)alloc((size_t)NCHUNKS * NHEADS * 64 * 128 * 4);
  float* Spre    = (float*)alloc((size_t)NCHUNKS * NHEADS * 64 * 128 * 4);
  float* AcumG   = (float*)alloc((size_t)NCHUNKS * NHEADS * 64 * 4);
  bf16*  hbf     = (bf16*)alloc((size_t)SEQ * D_MODEL * 2);
  bf16*  h2bf    = (bf16*)alloc((size_t)SEQ * D_INNER * 2);
  bf16*  xfbf    = (bf16*)alloc((size_t)SEQ * D_MODEL * 2);
  bf16*  embbf   = (bf16*)alloc((size_t)VOCAB_PAD * D_MODEL * 2);
  bf16*  winbf   = (bf16*)alloc((size_t)N_LAYER * D_MODEL * D_IN_PROJ_PAD * 2);
  bf16*  woutbf  = (bf16*)alloc((size_t)N_LAYER * D_INNER * D_MODEL * 2);
  float* logits  = (float*)d_out;

  // Weight conversions (padded bf16 copies)
  convert_pad_kernel<<<cdiv((long)VOCAB_PAD * D_MODEL, 256), 256, 0, stream>>>(
      emb, embbf, VOCAB, D_MODEL, VOCAB_PAD, D_MODEL);
  for (int l = 0; l < N_LAYER; ++l) {
    convert_pad_kernel<<<cdiv((long)D_MODEL * D_IN_PROJ_PAD, 256), 256, 0, stream>>>(
        W_in + (size_t)l * D_MODEL * D_IN_PROJ,
        winbf + (size_t)l * D_MODEL * D_IN_PROJ_PAD, D_MODEL, D_IN_PROJ, D_MODEL,
        D_IN_PROJ_PAD);
    convert_pad_kernel<<<cdiv((long)D_INNER * D_MODEL, 256), 256, 0, stream>>>(
        W_out + (size_t)l * D_INNER * D_MODEL, woutbf + (size_t)l * D_INNER * D_MODEL,
        D_INNER, D_MODEL, D_INNER, D_MODEL);
  }

  embed_kernel<<<cdiv((long)SEQ * D_MODEL, 256), 256, 0, stream>>>(ids, emb, x);

  for (int l = 0; l < N_LAYER; ++l) {
    const float* ln   = ln_w + (size_t)l * D_MODEL;
    const bf16*  Wi   = winbf + (size_t)l * D_MODEL * D_IN_PROJ_PAD;
    const bf16*  Wo   = woutbf + (size_t)l * D_INNER * D_MODEL;
    const float* cw   = conv_w + (size_t)l * CONV_CH * 4;
    const float* cb   = conv_b + (size_t)l * CONV_CH;
    const float* dtbs = dt_bias + (size_t)l * NHEADS;
    const float* Al   = A_log + (size_t)l * NHEADS;
    const float* Dl   = D_skip + (size_t)l * NHEADS;
    const float* nw   = norm_w + (size_t)l * D_INNER;

    rmsnorm_bf16_kernel<<<SEQ, 256, 0, stream>>>(x, ln, hbf, D_MODEL);
    gemm_bf16_kernel<0, 0><<<dim3(SEQ / 128, D_IN_PROJ_PAD / 64), 256, 0, stream>>>(
        hbf, Wi, zx, D_MODEL, D_IN_PROJ_PAD, D_IN_PROJ_PAD, D_IN_PROJ_PAD);
    dt_kernel<<<cdiv((long)SEQ * NHEADS, 256), 256, 0, stream>>>(zx, dtbs, dtb);
    conv_kernel<<<cdiv((long)SEQ * CONV_CH, 256), 256, 0, stream>>>(zx, cw, cb, xBCc);
    ssd_chunk_kernel<<<dim3(NCHUNKS, NHEADS), 256, 0, stream>>>(xBCc, dtb, Al, Y,
                                                                states, AcumG);
    ssd_scan_kernel<<<NHEADS, 256, 0, stream>>>(states, AcumG, Spre);
    ssd_yoff_kernel<<<dim3(NCHUNKS, NHEADS), 256, 0, stream>>>(xBCc, Spre, AcumG, Y);
    gated_norm_kernel<<<SEQ, 256, 0, stream>>>(Y, xBCc, zx, Dl, nw, h2bf);
    gemm_bf16_kernel<0, 0><<<dim3(SEQ / 128, D_MODEL / 64), 256, 0, stream>>>(
        h2bf, Wo, O, D_INNER, D_MODEL, D_MODEL, D_MODEL);
    add_kernel<<<cdiv((long)SEQ * D_MODEL, 256), 256, 0, stream>>>(x, O,
                                                                   SEQ * D_MODEL);
  }

  rmsnorm_bf16_kernel<<<SEQ, 256, 0, stream>>>(x, final_ln_w, xfbf, D_MODEL);
  gemm_bf16_kernel<1, 1><<<dim3(SEQ / 128, VOCAB_PAD / 64), 256, 0, stream>>>(
      xfbf, embbf, logits, D_MODEL, 0, VOCAB, VOCAB);
}
